// Linear8bit_10548439679180
// MI455X (gfx1250) — matrix operand order repaired and checked by reading, not compiled, not run on hardware
//
#include <hip/hip_runtime.h>

typedef _Float16 v16h __attribute__((ext_vector_type(16)));
typedef _Float16 v8h  __attribute__((ext_vector_type(8)));
typedef float    v8f  __attribute__((ext_vector_type(8)));
typedef int      v4i  __attribute__((ext_vector_type(4)));

#define MDIM 64
#define KDIM 8192
#define NDIM 8192
#define KSPLIT 8
#define KSLICE (KDIM / KSPLIT)           // 1024 per K-slice
#define KCHUNK 32                        // K per WMMA step
#define NITER (KSLICE / KCHUNK)          // 32 iterations
#define WAVES_PER_WG 8
#define NT_PER_WG (WAVES_PER_WG * 16)    // 128 N-columns per workgroup
#define ROW_V8H 5                        // 40 halfs/row (32 data + pad): 16B-aligned, bank-rotated
#define LDS_BUF (MDIM * ROW_V8H)         // one 64x32 fp16 A-chunk (5 KB)

// Workgroup barrier that only drains DScnt (not LOADcnt), so the NT weight
// prefetches issued 2 iterations ahead stay in flight across the barrier.
// Safe: all LDS traffic of a wave is complete when it signals, and the LDS
// A-buffer is double-buffered (reuse is two barriers apart).
__device__ __forceinline__ void wg_barrier_lds_only() {
    asm volatile(
        "s_wait_dscnt 0x0\n"
        "s_barrier_signal -1\n"
        "s_barrier_wait -1" ::: "memory");
}

// GEMM: each wave computes a 64(M) x 16(N) fp32 partial tile over one K-slice.
// Software-pipelined: A-stage 1 iter ahead (double-buffered LDS), B weights 2 iters ahead.
__global__ __launch_bounds__(256)
void Linear8bit_gemm_wmma(const _Float16* __restrict__ x,    // [64, 8192] fp16
                          const int*      __restrict__ wq,   // [8192, 8192] int8-in-int32
                          float*          __restrict__ ws)   // [KSPLIT, 64, 8192] partials
{
    __shared__ v8h lds_x[2 * LDS_BUF];                       // 10 KB, double-buffered

    const int tid  = threadIdx.x;
    const int wave = tid >> 5;
    const int lane = tid & 31;
    const int nblk = blockIdx.x >> 3;                        // / KSPLIT
    const int kslc = blockIdx.x & (KSPLIT - 1);

    const int nbase = nblk * NT_PER_WG + wave * 16;
    const int k0    = kslc * KSLICE;

    // B-fragment: lane<16 -> col N=lane, K 0..15; lane>=16 -> col N=lane-16, K 16..31
    const int bcol  = lane & 15;
    const int bk    = (lane >> 4) * 16;
    const int* wrow = wq + (size_t)(nbase + bcol) * KDIM + k0 + bk;

    // Cooperative A staging: thread t loads 8 halfs of row t/4, sub-chunk t%4
    const int arow   = tid >> 2;
    const int aslot  = tid & 3;
    const _Float16* xsrc = x + (size_t)arow * KDIM + k0 + aslot * 8;
    const int astore = arow * ROW_V8H + aslot;

    // A-fragment addressing (ISA 7.12.2, 16-bit A 16x32)
    const int fr = lane & 15;
    const int fs = lane >> 4;

    v8f acc[4] = {v8f{}, v8f{}, v8f{}, v8f{}};

    // ---- pipeline prologue: A chunk 0, B chunks 0 and 1 ----
    v8h a_stage = *(const v8h*)xsrc;
    v4i qa[4], qb[4];
    #pragma unroll
    for (int j = 0; j < 4; ++j)
        qa[j] = __builtin_nontemporal_load((const v4i*)wrow + j);
    #pragma unroll
    for (int j = 0; j < 4; ++j)
        qb[j] = __builtin_nontemporal_load((const v4i*)(wrow + KCHUNK) + j);

    for (int it = 0; it < NITER; ++it) {
        const int buf = (it & 1) * LDS_BUF;

        // stage A chunk 'it' (fetched one iteration ago)
        lds_x[buf + astore] = a_stage;

        // issue loads: A for it+1, B for it+2 (clamped tail -> branch-free, in-bounds)
        const int kkA = ((it + 1 < NITER) ? (it + 1) : (NITER - 1)) * KCHUNK;
        a_stage = *(const v8h*)(xsrc + kkA);
        const int kkB = ((it + 2 < NITER) ? (it + 2) : (NITER - 1)) * KCHUNK;
        v4i qn[4];
        #pragma unroll
        for (int j = 0; j < 4; ++j)
            qn[j] = __builtin_nontemporal_load((const v4i*)(wrow + kkB) + j);

        // convert current B tile: int32 -> fp16 (co-executes with XDL WMMAs)
        v16h bfrag;
        #pragma unroll
        for (int g = 0; g < 4; ++g) {
            #pragma unroll
            for (int j = 0; j < 4; ++j)
                bfrag[g * 4 + j] = (_Float16)qa[g][j];
        }
        #pragma unroll
        for (int j = 0; j < 4; ++j) { qa[j] = qb[j]; qb[j] = qn[j]; }

        wg_barrier_lds_only();           // A chunk 'it' visible; global prefetches stay in flight

        // all 4 A fragments up front: one dscnt wait, then 4 back-to-back WMMAs
        v16h af[4];
        #pragma unroll
        for (int mt = 0; mt < 4; ++mt) {
            const int row = mt * 16 + fr;
            v8h lo = lds_x[buf + row * ROW_V8H + 2 * fs];        // K = fs*16 .. +7
            v8h hi = lds_x[buf + row * ROW_V8H + 2 * fs + 1];    // K = fs*16 + 8 .. +7
            af[mt] = __builtin_shufflevector(lo, hi,
                0,1,2,3,4,5,6,7,8,9,10,11,12,13,14,15);
        }
        #pragma unroll
        for (int mt = 0; mt < 4; ++mt)
            acc[mt] = __builtin_amdgcn_wmma_f32_16x16x32_f16(
                false, af[mt], false, bfrag, (short)0, acc[mt], false, false);

#if defined(__has_builtin)
#if __has_builtin(__builtin_amdgcn_sched_group_barrier)
        // Pin schedule: 8 DS-reads first, then 4 WMMAs back-to-back.
        __builtin_amdgcn_sched_group_barrier(0x100, 8, 0);   // DS read x8
        __builtin_amdgcn_sched_group_barrier(0x008, 4, 0);   // MFMA/WMMA x4
#endif
#endif
    }

    // Partial tile out (C/D layout: VGPR r -> M = r + 8*(lane>=16), N = lane&15)
    const int nc = nbase + (lane & 15);
    const int mo = (lane >> 4) * 8;
    float* wbase = ws + (size_t)kslc * ((size_t)MDIM * NDIM);
    #pragma unroll
    for (int mt = 0; mt < 4; ++mt) {
        #pragma unroll
        for (int r = 0; r < 8; ++r)
            wbase[(size_t)(mt * 16 + mo + r) * NDIM + nc] = acc[mt][r];
    }
}

// Deterministic fixed-order split-K reduction + rowwise dequant scale + bias.
__global__ __launch_bounds__(256)
void Linear8bit_reduce(const float* __restrict__ ws,
                       const _Float16* __restrict__ scales,
                       const _Float16* __restrict__ bias,
                       float* __restrict__ out)
{
    const int i = blockIdx.x * 256 + threadIdx.x;            // 0 .. 64*8192-1
    const int n = i & (NDIM - 1);
    const size_t MN = (size_t)MDIM * NDIM;
    float s = 0.0f;
    #pragma unroll
    for (int p = 0; p < KSPLIT; ++p)
        s += ws[(size_t)p * MN + i];
    float sc = (float)(_Float16)(scales[n] / (_Float16)127.0f);  // fp16 scale/127 like reference
    out[i] = s * sc + (float)bias[n];
}

extern "C" void kernel_launch(void* const* d_in, const int* in_sizes, int n_in,
                              void* d_out, int out_size, void* d_ws, size_t ws_size,
                              hipStream_t stream) {
    const _Float16* x      = (const _Float16*)d_in[0];
    const int*      wq     = (const int*)d_in[1];
    const _Float16* scales = (const _Float16*)d_in[2];
    const _Float16* bias   = (const _Float16*)d_in[3];
    float*          out    = (float*)d_out;
    float*          ws     = (float*)d_ws;   // needs KSPLIT*64*8192*4 = 16 MB

    const int gemm_blocks = (NDIM / NT_PER_WG) * KSPLIT;     // 64 * 8 = 512 workgroups
    Linear8bit_gemm_wmma<<<gemm_blocks, 256, 0, stream>>>(x, wq, ws);

    const int total = MDIM * NDIM;                           // 524288
    Linear8bit_reduce<<<total / 256, 256, 0, stream>>>(ws, scales, bias, out);
}
